// DisasterMoEModel_20229295964549
// MI455X (gfx1250) — matrix operand order repaired and checked by reading, not compiled, not run on hardware
//
#include <hip/hip_runtime.h>
#include <hip/hip_bf16.h>

typedef _Float16 h8  __attribute__((ext_vector_type(8)));
typedef _Float16 h16 __attribute__((ext_vector_type(16)));
typedef float    f8  __attribute__((ext_vector_type(8)));

// ---- fp16 weight arena layout inside d_ws (offsets in halves) ----
#define OFF_ENC1   0        // [128][64]
#define OFF_ENC2   8192     // [64][128]
#define OFF_FE     16384    // [64][64]
#define OFF_ME     20480    // [64][32]  (K padded 22->32, zero fill)
#define OFF_FP     22528    // [64][64]
#define OFF_ATTV   26624    // [64][64]
#define OFF_ATTO   30720    // [64][64]
#define OFF_G1     34816    // [64][128]
#define OFF_EW1    43008    // [5][128][128]
#define OFF_EW2    124928   // [5][16][128] (N padded 10->16, zero fill)
#define W16_TOTAL  135168
#define ACC_BYTE_OFF (W16_TOTAL * 2)   // 8 floats of gate-prob accumulators

#define NW 4   // waves (16-sample tiles) per block

__device__ __forceinline__ void wsync() {
    __builtin_amdgcn_wave_barrier();
    asm volatile("s_wait_dscnt 0" ::: "memory");
    __builtin_amdgcn_wave_barrier();
}

// ---------------- weight prep: fp32 -> fp16 with padding ----------------
__global__ void prep_weights(const float* enc_W1, const float* enc_W2, const float* fe_W,
                             const float* me_W, const float* fp_W, const float* att_vW,
                             const float* att_oW, const float* g1_W, const float* e_W1,
                             const float* e_W2, _Float16* w16, float* acc) {
    int tid = blockIdx.x * blockDim.x + threadIdx.x;
    if (tid < 8) acc[tid] = 0.0f;
    for (int i = tid; i < W16_TOTAL; i += gridDim.x * blockDim.x) {
        float v;
        if      (i < 8192)   v = enc_W1[i];
        else if (i < 16384)  v = enc_W2[i - 8192];
        else if (i < 20480)  v = fe_W[i - 16384];
        else if (i < 22528) { int j = i - 20480; int n = j >> 5, k = j & 31;
                              v = (k < 22) ? me_W[n * 22 + k] : 0.0f; }
        else if (i < 26624)  v = fp_W[i - 22528];
        else if (i < 30720)  v = att_vW[i - 26624];
        else if (i < 34816)  v = att_oW[i - 30720];
        else if (i < 43008)  v = g1_W[i - 34816];
        else if (i < 124928) v = e_W1[i - 43008];
        else { int j = i - 124928; int e = j >> 11, r = (j >> 7) & 15, k = j & 127;
               v = (r < 10) ? e_W2[(e * 10 + r) * 128 + k] : 0.0f; }
        w16[i] = (_Float16)v;
    }
}

// ---------------- WMMA GEMM over a 16-row tile ----------------
// A: LDS fp16, row-major [16][strideA].  W: global fp16, n-major [N][strideW].
// C: LDS fp32, row-major [16][strideC].  Computes C = A * W^T (+bias) (+=C if accum).
__device__ __forceinline__ void gemm(const _Float16* A, int strideA,
                                     const _Float16* W, int strideW,
                                     int Kloc, int N, const float* bias, int biasN,
                                     float* C, int strideC, bool accum, int lane) {
    const int col  = lane & 15;
    const int r0   = (lane & 16) ? 8 : 0;
    const int aoff = (lane & 16) ? 16 : 0;   // contiguous k order, same as B: hi-half lanes hold K+16..31
    const int boff = (lane & 16) ? 16 : 0;   // contiguous k order, same as A: hi-half lanes hold K+16..31
    for (int nt = 0; nt < (N >> 4); ++nt) {
        f8 acc;
        if (accum) {
#pragma unroll
            for (int r = 0; r < 8; ++r) acc[r] = C[(r0 + r) * strideC + (nt << 4) + col];
        } else {
#pragma unroll
            for (int r = 0; r < 8; ++r) acc[r] = 0.0f;
        }
        for (int kt = 0; kt < (Kloc >> 5); ++kt) {
            const _Float16* pa = A + col * strideA + (kt << 5) + aoff;
            h8 lo = *(const h8*)pa;
            h8 hi = *(const h8*)(pa + 8);
            h16 a;
#pragma unroll
            for (int i = 0; i < 8; ++i) { a[i] = lo[i]; a[i + 8] = hi[i]; }
            const _Float16* pb = W + ((nt << 4) + col) * strideW + (kt << 5) + boff;
            h16 b = *(const h16*)pb;
            acc = __builtin_amdgcn_wmma_f32_16x16x32_f16(false, a, false, b,
                                                         (short)0, acc, false, false);
        }
        float bv = 0.0f;
        if (bias) { int n = (nt << 4) + col; bv = (n < biasN) ? bias[n] : 0.0f; }
#pragma unroll
        for (int r = 0; r < 8; ++r) C[(r0 + r) * strideC + (nt << 4) + col] = acc[r] + bv;
    }
}

// LayerNorm + exact GELU, fp32 in -> fp16 out. 2 lanes per row.
__device__ __forceinline__ void ln_gelu(const float* C, int H, int strideC,
                                        const float* g, const float* b,
                                        _Float16* dst, int strideD, int lane) {
    int m = lane & 15, half = lane >> 4, h2 = H >> 1, c0 = half * h2;
    float s = 0.0f, ss = 0.0f;
    for (int c = 0; c < h2; ++c) { float v = C[m * strideC + c0 + c]; s += v; ss += v * v; }
    s  += __shfl_xor(s, 16, 32);
    ss += __shfl_xor(ss, 16, 32);
    float mean = s / (float)H;
    float var  = ss / (float)H - mean * mean;
    float inv  = rsqrtf(var + 1e-5f);
    for (int c = 0; c < h2; ++c) {
        int col = c0 + c;
        float v  = (C[m * strideC + col] - mean) * inv * g[col] + b[col];
        float ge = 0.5f * v * (1.0f + erff(v * 0.70710678118f));
        dst[m * strideD + col] = (_Float16)ge;
    }
}

__device__ __forceinline__ void conv16(const float* C, int H, int strideC,
                                       _Float16* dst, int strideD, int lane) {
    for (int i = lane; i < 16 * H; i += 32) {
        int m = i / H, c = i - m * H;
        dst[m * strideD + c] = (_Float16)C[m * strideC + c];
    }
}

struct Params {
    const float* x; const int* dtype; const float* sev; const float* loc;
    const float *enc_b1, *enc_lng, *enc_lnb, *enc_b2;
    const float* emb;
    const float *fe_b, *fe_lng, *fe_lnb, *me_b, *me_lng, *me_lnb;
    const float *fp_b, *att_vb, *att_ob;
    const float *g1_b, *g_lng, *g_lnb, *g2_W, *g2_b;
    const float *e_b1, *e_lng, *e_lnb, *e_b2;
    const float *h_sev_W, *h_sev_b, *h_spr_W, *h_spr_b, *h_imp_W, *h_imp_b;
    const float *h_res_W, *h_res_b, *h_ano_W, *h_ano_b;
    const _Float16* w16;
    float* out; float* acc;
    int Bn;
};

__global__ __launch_bounds__(128) void moe_main(Params p) {
    __shared__ __align__(128) float     sC[NW][2048];   // 16x128 fp32 per wave
    __shared__ __align__(128) _Float16  sE[NW][2048];   // expert input [enc | meta_enc] fp16
    __shared__ __align__(128) _Float16  sP[NW][2048];   // fp16 staging
    const int lane = threadIdx.x & 31;
    const int wv   = threadIdx.x >> 5;
    const int tile = blockIdx.x * NW + wv;
    if (tile * 16 >= p.Bn) return;
    float*     C  = sC[wv];
    _Float16*  E  = sE[wv];
    _Float16*  Pb = sP[wv];
    const int row0 = tile * 16;
    const _Float16* W = p.w16;

    // x tile -> fp16 (stride 64)
    for (int i = lane; i < 1024; i += 32) {
        int m = i >> 6, c = i & 63;
        Pb[(m << 6) + c] = (_Float16)p.x[(row0 + m) * 64 + c];
    }
    wsync();
    // encoder: Linear(64,128) + LN + GELU + Linear(128,64)
    gemm(Pb, 64, W + OFF_ENC1, 64, 64, 128, p.enc_b1, 128, C, 128, false, lane);
    wsync();
    ln_gelu(C, 128, 128, p.enc_lng, p.enc_lnb, Pb, 128, lane);
    wsync();
    gemm(Pb, 128, W + OFF_ENC2, 128, 128, 64, p.enc_b2, 64, C, 64, false, lane);
    wsync();
    conv16(C, 64, 64, E, 128, lane);            // encoded -> E[:, 0:64]
    wsync();
    // meta = [emb(type) | severity | location | 0pad] -> 16x32 fp16
    for (int i = lane; i < 512; i += 32) {
        int m = i >> 5, c = i & 31, r = row0 + m;
        float v;
        if      (c < 16) v = p.emb[p.dtype[r] * 16 + c];
        else if (c < 20) v = p.sev[r * 4 + (c - 16)];
        else if (c < 22) v = p.loc[r * 2 + (c - 20)];
        else             v = 0.0f;
        Pb[(m << 5) + c] = (_Float16)v;
    }
    wsync();
    gemm(Pb, 32, W + OFF_ME, 32, 32, 64, p.me_b, 64, C, 64, false, lane);
    wsync();
    ln_gelu(C, 64, 64, p.me_lng, p.me_lnb, E + 64, 128, lane);   // meta_enc -> E[:, 64:128]
    wsync();
    // feat = GELU(LN(encoded @ fe))
    gemm(E, 128, W + OFF_FE, 64, 64, 64, p.fe_b, 64, C, 64, false, lane);
    wsync();
    ln_gelu(C, 64, 64, p.fe_lng, p.fe_lnb, Pb, 64, lane);        // feat in Pb[0:1024]
    wsync();
    // attended = out_proj(v_proj(feat @ fp + b))   (len-1 softmax == 1; cp_W elided)
    gemm(Pb, 64, W + OFF_FP, 64, 64, 64, p.fp_b, 64, C, 64, false, lane);
    wsync();
    conv16(C, 64, 64, Pb + 1024, 64, lane);                      // kv
    wsync();
    gemm(Pb + 1024, 64, W + OFF_ATTV, 64, 64, 64, p.att_vb, 64, C, 64, false, lane);
    wsync();
    conv16(C, 64, 64, Pb, 64, lane);                             // v (feat dead)
    wsync();
    gemm(Pb, 64, W + OFF_ATTO, 64, 64, 64, p.att_ob, 64, C, 64, false, lane);
    wsync();
    conv16(C, 64, 64, Pb + 1024, 64, lane);                      // attended fp16
    wsync();
    // gate: GELU(LN([attended|meta_enc] @ g1)) @ g2  — split-K accumulation
    gemm(Pb + 1024, 64, W + OFF_G1,      128, 64, 64, nullptr, 0,  C, 64, false, lane);
    gemm(E + 64,    128, W + OFF_G1 + 64, 128, 64, 64, p.g1_b, 64, C, 64, true,  lane);
    wsync();
    ln_gelu(C, 64, 64, p.g_lng, p.g_lnb, Pb, 64, lane);          // G in Pb (stride 64)
    wsync();

    // g2 (64->5) + top-2 routing, in fp32 VALU (2 lanes per row)
    const int m = lane & 15, half = lane >> 4;
    float lg[5];
#pragma unroll
    for (int e = 0; e < 5; ++e) {
        float s = 0.0f;
        for (int k = 0; k < 32; ++k) {
            int c = half * 32 + k;
            s += (float)Pb[(m << 6) + c] * p.g2_W[e * 64 + c];
        }
        s += __shfl_xor(s, 16, 32);
        lg[e] = s + p.g2_b[e];
    }
    int i1 = 0;
#pragma unroll
    for (int e = 1; e < 5; ++e) if (lg[e] > lg[i1]) i1 = e;
    int i2 = (i1 == 0) ? 1 : 0;
#pragma unroll
    for (int e = 0; e < 5; ++e) if (e != i1 && lg[e] > lg[i2]) i2 = e;
    float e2 = expf(lg[i2] - lg[i1]);
    float w1 = 1.0f / (1.0f + e2), w2 = e2 / (1.0f + e2);
    float mk[5];
#pragma unroll
    for (int e = 0; e < 5; ++e) mk[e] = (e == i1) ? w1 : ((e == i2) ? w2 : 0.0f);
    // full softmax of logits for load-balance term
    {
        float mx = lg[0];
#pragma unroll
        for (int e = 1; e < 5; ++e) mx = fmaxf(mx, lg[e]);
        float pe[5], ssum = 0.0f;
#pragma unroll
        for (int e = 0; e < 5; ++e) { pe[e] = expf(lg[e] - mx); ssum += pe[e]; }
        float inv = 1.0f / ssum;
#pragma unroll
        for (int e = 0; e < 5; ++e) {
            float q = pe[e] * inv;
            q += __shfl_xor(q, 1, 32);  q += __shfl_xor(q, 2, 32);
            q += __shfl_xor(q, 4, 32);  q += __shfl_xor(q, 8, 32);
            q += __shfl_xor(q, 16, 32);            // sums 16 rows twice
            if (lane == 0) atomicAdd(p.acc + e, q * 0.5f);
        }
    }

    // experts: Linear(128,128)+LN+GELU+Linear(128,out), heads, routing mask
    for (int e = 0; e < 5; ++e) {
        wsync();
        gemm(E, 128, W + OFF_EW1 + e * 16384, 128, 128, 128,
             p.e_b1 + e * 128, 128, C, 128, false, lane);
        wsync();
        ln_gelu(C, 128, 128, p.e_lng + e * 128, p.e_lnb + e * 128, Pb, 128, lane);
        wsync();
        gemm(Pb, 128, W + OFF_EW2 + e * 2048, 128, 128, 16,
             p.e_b2 + e * 10, 10, C, 16, false, lane);
        wsync();
        if (lane < 16) {
            const float* o = C + lane * 16;
            float* outp = p.out + (size_t)(row0 + lane) * 20;
            if (e == 0) {                                   // softmax4 @ h_sev
                float v[4], mx = o[0], s = 0.0f;
                for (int c = 1; c < 4; ++c) mx = fmaxf(mx, o[c]);
                for (int c = 0; c < 4; ++c) { v[c] = expf(o[c] - mx); s += v[c]; }
                for (int c = 0; c < 4; ++c) v[c] /= s;
                for (int c = 0; c < 4; ++c) {
                    float r = p.h_sev_b[c];
                    for (int j = 0; j < 4; ++j) r += p.h_sev_W[c * 4 + j] * v[j];
                    outp[c] = r * mk[0];
                }
            } else if (e == 1) {                            // softplus3 @ h_spr
                float v[3];
                for (int j = 0; j < 3; ++j) v[j] = (o[j] > 20.0f) ? o[j] : log1pf(expf(o[j]));
                for (int c = 0; c < 3; ++c) {
                    float r = p.h_spr_b[c];
                    for (int j = 0; j < 3; ++j) r += p.h_spr_W[c * 3 + j] * v[j];
                    outp[4 + c] = r * mk[1];
                }
            } else if (e == 2) {                            // softplus2 @ h_imp
                float v[2];
                for (int j = 0; j < 2; ++j) v[j] = (o[j] > 20.0f) ? o[j] : log1pf(expf(o[j]));
                for (int c = 0; c < 2; ++c) {
                    float r = p.h_imp_b[c];
                    for (int j = 0; j < 2; ++j) r += p.h_imp_W[c * 2 + j] * v[j];
                    outp[7 + c] = r * mk[2];
                }
            } else if (e == 3) {                            // softmax10 @ h_res
                float v[10], mx = o[0], s = 0.0f;
                for (int c = 1; c < 10; ++c) mx = fmaxf(mx, o[c]);
                for (int c = 0; c < 10; ++c) { v[c] = expf(o[c] - mx); s += v[c]; }
                for (int c = 0; c < 10; ++c) v[c] /= s;
                for (int c = 0; c < 10; ++c) {
                    float r = p.h_res_b[c];
                    for (int j = 0; j < 10; ++j) r += p.h_res_W[c * 10 + j] * v[j];
                    outp[9 + c] = r * mk[3];
                }
            } else {                                        // sigmoid @ h_ano
                float sg = 1.0f / (1.0f + expf(-o[0]));
                outp[19] = (p.h_ano_W[0] * sg + p.h_ano_b[0]) * mk[4];
            }
        }
    }
}

__global__ void lb_finalize(const float* acc, float* out, int Bn, int idx) {
    float eu[5], mx = -1e30f;
    for (int e = 0; e < 5; ++e) { eu[e] = acc[e] / (float)Bn; mx = fmaxf(mx, eu[e]); }
    float s = 0.0f;
    for (int e = 0; e < 5; ++e) s += expf(eu[e] - mx);
    float lse = mx + logf(s);
    float t = 0.2f, lt = logf(t), lb = 0.0f;
    for (int e = 0; e < 5; ++e) lb += t * (lt - (eu[e] - lse));
    out[idx] = lb * 0.2f;
}

extern "C" void kernel_launch(void* const* d_in, const int* in_sizes, int n_in,
                              void* d_out, int out_size, void* d_ws, size_t ws_size,
                              hipStream_t stream) {
    const float* x       = (const float*)d_in[0];
    const int*   dtype   = (const int*)d_in[1];
    const float* sev     = (const float*)d_in[2];
    const float* loc     = (const float*)d_in[3];
    const float* enc_W1  = (const float*)d_in[4];
    const float* enc_b1  = (const float*)d_in[5];
    const float* enc_lng = (const float*)d_in[6];
    const float* enc_lnb = (const float*)d_in[7];
    const float* enc_W2  = (const float*)d_in[8];
    const float* enc_b2  = (const float*)d_in[9];
    const float* emb     = (const float*)d_in[10];
    const float* fe_W    = (const float*)d_in[11];
    const float* fe_b    = (const float*)d_in[12];
    const float* fe_lng  = (const float*)d_in[13];
    const float* fe_lnb  = (const float*)d_in[14];
    const float* me_W    = (const float*)d_in[15];
    const float* me_b    = (const float*)d_in[16];
    const float* me_lng  = (const float*)d_in[17];
    const float* me_lnb  = (const float*)d_in[18];
    const float* fp_W    = (const float*)d_in[21];
    const float* fp_b    = (const float*)d_in[22];
    const float* att_vW  = (const float*)d_in[23];
    const float* att_vb  = (const float*)d_in[24];
    const float* att_oW  = (const float*)d_in[25];
    const float* att_ob  = (const float*)d_in[26];
    const float* g1_W    = (const float*)d_in[27];
    const float* g1_b    = (const float*)d_in[28];
    const float* g_lng   = (const float*)d_in[29];
    const float* g_lnb   = (const float*)d_in[30];
    const float* g2_W    = (const float*)d_in[31];
    const float* g2_b    = (const float*)d_in[32];
    const float* e_W1    = (const float*)d_in[33];
    const float* e_b1    = (const float*)d_in[34];
    const float* e_lng   = (const float*)d_in[35];
    const float* e_lnb   = (const float*)d_in[36];
    const float* e_W2    = (const float*)d_in[37];
    const float* e_b2    = (const float*)d_in[38];

    _Float16* w16 = (_Float16*)d_ws;
    float* acc = (float*)((char*)d_ws + ACC_BYTE_OFF);
    int Bn = in_sizes[0] / 64;

    prep_weights<<<256, 256, 0, stream>>>(enc_W1, enc_W2, fe_W, me_W, fp_W,
                                          att_vW, att_oW, g1_W, e_W1, e_W2, w16, acc);

    Params P;
    P.x = x; P.dtype = dtype; P.sev = sev; P.loc = loc;
    P.enc_b1 = enc_b1; P.enc_lng = enc_lng; P.enc_lnb = enc_lnb; P.enc_b2 = enc_b2;
    P.emb = emb;
    P.fe_b = fe_b; P.fe_lng = fe_lng; P.fe_lnb = fe_lnb;
    P.me_b = me_b; P.me_lng = me_lng; P.me_lnb = me_lnb;
    P.fp_b = fp_b; P.att_vb = att_vb; P.att_ob = att_ob;
    P.g1_b = g1_b; P.g_lng = g_lng; P.g_lnb = g_lnb; P.g2_W = g2_W; P.g2_b = g2_b;
    P.e_b1 = e_b1; P.e_lng = e_lng; P.e_lnb = e_lnb; P.e_b2 = e_b2;
    P.h_sev_W = (const float*)d_in[39]; P.h_sev_b = (const float*)d_in[40];
    P.h_spr_W = (const float*)d_in[41]; P.h_spr_b = (const float*)d_in[42];
    P.h_imp_W = (const float*)d_in[43]; P.h_imp_b = (const float*)d_in[44];
    P.h_res_W = (const float*)d_in[45]; P.h_res_b = (const float*)d_in[46];
    P.h_ano_W = (const float*)d_in[47]; P.h_ano_b = (const float*)d_in[48];
    P.w16 = w16; P.out = (float*)d_out; P.acc = acc; P.Bn = Bn;

    int tiles  = Bn / 16;
    int blocks = (tiles + NW - 1) / NW;
    moe_main<<<blocks, 128, 0, stream>>>(P);

    lb_finalize<<<1, 1, 0, stream>>>(acc, (float*)d_out, Bn, out_size - 1);
}
